// MultiAngularPenaltySMLoss_8993661517958
// MI455X (gfx1250) — matrix operand order repaired and checked by reading, not compiled, not run on hardware
//
#include <hip/hip_runtime.h>
#include <hip/hip_bf16.h>
#include <cstdint>
#include <cstddef>

// Problem constants (from reference)
#define IN_F   512
#define OUT_F  50000
#define NBATCH 512
#define NE     3
#define S_SCALE 64.0f
#define M_MARG  0.5f
#define EPS_C   1e-7f

typedef __attribute__((ext_vector_type(16))) _Float16 v16h;
typedef __attribute__((ext_vector_type(8)))  float    v8f;

union Frag16 {
    v16h v;
    _Float16 h[16];
    uint4 q[2];
};

__device__ __forceinline__ v8f wmma_f16(v16h a, v16h b, v8f c) {
    // D(16x16,f32) = A(16x32,f16) * B(32x16,f16) + C
    return __builtin_amdgcn_wmma_f32_16x16x32_f16(
        /*neg_a=*/false, a, /*neg_b=*/false, b,
        /*c_mod=*/(short)0, c, /*reuse_a=*/false, /*reuse_b=*/false);
}

// ---------------------------------------------------------------------------
// Kernel 1: row-normalize x (512x512 f32) -> xn f16
// ---------------------------------------------------------------------------
__global__ void k_normalize(const float* __restrict__ x,
                            _Float16* __restrict__ xn) {
    __shared__ float lds[8];
    const int row = blockIdx.x;
    const float* xr = x + (size_t)row * IN_F;
    float a = xr[threadIdx.x];
    float b = xr[threadIdx.x + 256];
    float ss = a * a + b * b;
    #pragma unroll
    for (int off = 16; off > 0; off >>= 1) ss += __shfl_down(ss, off, 32);
    const int lane = threadIdx.x & 31, wid = threadIdx.x >> 5;
    if (lane == 0) lds[wid] = ss;
    __syncthreads();
    float tot = 0.f;
    #pragma unroll
    for (int i = 0; i < 8; ++i) tot += lds[i];
    const float inv = rsqrtf(tot);
    _Float16* xo = xn + (size_t)row * IN_F;
    xo[threadIdx.x]       = (_Float16)(a * inv);
    xo[threadIdx.x + 256] = (_Float16)(b * inv);
}

// ---------------------------------------------------------------------------
// Kernel 2: wf[n,o] = max_e  xn[n,:] . W[e,o,:]   via f16 WMMA, f32 accum
// Block tile: 128 N x 128 O; 8 waves; each wave owns a distinct 16-O strip
// across all 128 N (8 WMMA tiles). W fp32->f16 conversion: 16 elem/lane
// per k-step feeding 8 WMMAs, no in-block duplication.
// k-loop kept rolled (unroll 1) to bound register pressure (no spills);
// launch_bounds(256,2) = one 8-wave block per WGP, ~512 VGPR headroom.
// grid = (4 N-blocks [fast-moving, share W via L2], 391 O-blocks)
// ---------------------------------------------------------------------------
__global__ void __launch_bounds__(256, 2)
k_gemm_max(const _Float16* __restrict__ xn,
           const float* __restrict__ W,
           float* __restrict__ wf) {
    const int tid  = threadIdx.x;
    const int lane = tid & 31;
    const int wv   = tid >> 5;                      // 0..7 -> O strip
    const int nbase = blockIdx.x * 128;             // grid.x = 4
    const int otile = blockIdx.y * 128 + wv * 16;   // grid.y = 391

    const int l15 = lane & 15;
    const int lhi = lane >> 4;                      // 0 or 1

    v8f vmax[8];
    #pragma unroll
    for (int t = 0; t < 8; ++t)
        #pragma unroll
        for (int r = 0; r < 8; ++r) vmax[t][r] = -3.0e38f;

    // A fragment source: 16-bit A 16x32 layout (ISA 7.12.2):
    //   lanes 0-15: K = k0+0..7 (elems 0-7) and k0+16..23 (elems 8-15)
    //   lanes16-31: K = k0+8..15 and k0+24..31
    // One base pointer; n-tile stride folded into immediate offsets.
    const _Float16* abase = xn + (size_t)(nbase + l15) * IN_F + lhi * 8;

    // B fragment source: 16-bit B 32x16 layout:
    //   lane j holds column o = otile + (j&15); lanes 0-15 -> K=k0+0..7,k0+16..23,
    //   lanes 16-31 -> K=k0+8..15,k0+24..31 (same order as A).
    int orow = otile + l15;
    if (orow > OUT_F - 1) orow = OUT_F - 1;

    #pragma unroll 1
    for (int e = 0; e < NE; ++e) {
        const float* bp = W + ((size_t)e * OUT_F + orow) * IN_F + lhi * 8;

        v8f cur[8];
        #pragma unroll
        for (int t = 0; t < 8; ++t)
            #pragma unroll
            for (int r = 0; r < 8; ++r) cur[t][r] = 0.f;

        #pragma unroll 1
        for (int ks = 0; ks < IN_F / 32; ++ks) {
            const int k0 = ks * 32;
            // Speculative prefetch of next K slice (silently dropped if OOB).
            __builtin_prefetch(bp + k0 + 32, 0, 0);   // global_prefetch_b8

            // Load+convert one 32-wide K slice of this wave's O strip.
            Frag16 b;
            #pragma unroll
            for (int c = 0; c < 4; ++c) {
                float4 f = *(const float4*)(bp + k0 + c * 4 + (c >> 1) * 8);
                b.h[c * 4 + 0] = (_Float16)f.x;
                b.h[c * 4 + 1] = (_Float16)f.y;
                b.h[c * 4 + 2] = (_Float16)f.z;
                b.h[c * 4 + 3] = (_Float16)f.w;
            }

            #pragma unroll
            for (int nt = 0; nt < 8; ++nt) {
                Frag16 a;
                a.q[0] = *(const uint4*)(abase + (size_t)nt * 16 * IN_F + k0);
                a.q[1] = *(const uint4*)(abase + (size_t)nt * 16 * IN_F + k0 + 16);
                cur[nt] = wmma_f16(a.v, b.v, cur[nt]);
            }
        }

        #pragma unroll
        for (int t = 0; t < 8; ++t)
            #pragma unroll
            for (int r = 0; r < 8; ++r)
                vmax[t][r] = fmaxf(vmax[t][r], cur[t][r]);
    }

    // D layout: element r of v8f -> M = r + 8*lhi, N = lane&15
    const int o = otile + l15;
    if (o < OUT_F) {
        #pragma unroll
        for (int nt = 0; nt < 8; ++nt) {
            float* dst = wf + (size_t)(nbase + nt * 16 + 8 * lhi) * OUT_F + o;
            #pragma unroll
            for (int r = 0; r < 8; ++r)
                dst[(size_t)r * OUT_F] = vmax[nt][r];
        }
    }
}

// ---------------------------------------------------------------------------
// Kernel 3: per-row loss term  L[n] = num - log(exp(num) + sum_o e^{S wf} - e^{S tgt})
// ---------------------------------------------------------------------------
__global__ void k_rowloss(const float* __restrict__ wf,
                          const int* __restrict__ labels,
                          float* __restrict__ wsL) {
    __shared__ float lds[8];
    const int n = blockIdx.x;
    const float* row = wf + (size_t)n * OUT_F;
    float s = 0.f;
    for (int o = threadIdx.x; o < OUT_F; o += 256)
        s += __expf(S_SCALE * row[o]);
    #pragma unroll
    for (int off = 16; off > 0; off >>= 1) s += __shfl_down(s, off, 32);
    const int lane = threadIdx.x & 31, wid = threadIdx.x >> 5;
    if (lane == 0) lds[wid] = s;
    __syncthreads();
    if (threadIdx.x == 0) {
        float tot = 0.f;
        #pragma unroll
        for (int i = 0; i < 8; ++i) tot += lds[i];
        const float tgt = row[labels[n]];
        const float tc  = fminf(fmaxf(tgt, -1.0f + EPS_C), 1.0f - EPS_C);
        const float num = S_SCALE * cosf(acosf(tc) + M_MARG);
        const float den = expf(num) + tot - expf(S_SCALE * tgt);
        wsL[n] = num - logf(den);
    }
}

// ---------------------------------------------------------------------------
// Kernel 4: Gram dots for inner loss. block b -> i = b/3, pair = b%3
// G = sum_o W[a,o,i] * W[b,o,i]  (stride IN_F)
// ---------------------------------------------------------------------------
__global__ void k_gram(const float* __restrict__ W, float* __restrict__ wsG) {
    __shared__ float lds[8];
    const int blk = blockIdx.x;           // 0..8
    const int i = blk / 3;
    const int p = blk % 3;
    const int pa = (p == 2) ? 1 : 0;      // pairs: (0,1),(0,2),(1,2)
    const int pb = (p == 0) ? 1 : 2;
    const float* Wa = W + (size_t)pa * OUT_F * IN_F + i;
    const float* Wb = W + (size_t)pb * OUT_F * IN_F + i;
    float s = 0.f;
    for (int o = threadIdx.x; o < OUT_F; o += 256)
        s += Wa[(size_t)o * IN_F] * Wb[(size_t)o * IN_F];
    #pragma unroll
    for (int off = 16; off > 0; off >>= 1) s += __shfl_down(s, off, 32);
    const int lane = threadIdx.x & 31, wid = threadIdx.x >> 5;
    if (lane == 0) lds[wid] = s;
    __syncthreads();
    if (threadIdx.x == 0) {
        float tot = 0.f;
        #pragma unroll
        for (int k = 0; k < 8; ++k) tot += lds[k];
        wsG[blk] = tot;
    }
}

// ---------------------------------------------------------------------------
// Kernel 5: finalize   loss = -mean(L) + sum_b acos(clip(G[b]))/9
// ---------------------------------------------------------------------------
__global__ void k_final(const float* __restrict__ wsL,
                        const float* __restrict__ wsG,
                        float* __restrict__ out_loss) {
    __shared__ float lds[8];
    float s = wsL[threadIdx.x] + wsL[threadIdx.x + 256];
    #pragma unroll
    for (int off = 16; off > 0; off >>= 1) s += __shfl_down(s, off, 32);
    const int lane = threadIdx.x & 31, wid = threadIdx.x >> 5;
    if (lane == 0) lds[wid] = s;
    __syncthreads();
    if (threadIdx.x == 0) {
        float tot = 0.f;
        #pragma unroll
        for (int i = 0; i < 8; ++i) tot += lds[i];
        float inner = 0.f;
        #pragma unroll
        for (int b = 0; b < 9; ++b) {
            float g = wsG[b];
            g = fminf(fmaxf(g, -1.0f + EPS_C), 1.0f - EPS_C);
            inner += acosf(g);
        }
        inner *= (1.0f / 9.0f);
        *out_loss = -(tot / (float)NBATCH) + inner;
    }
}

// ---------------------------------------------------------------------------
extern "C" void kernel_launch(void* const* d_in, const int* in_sizes, int n_in,
                              void* d_out, int out_size, void* d_ws, size_t ws_size,
                              hipStream_t stream) {
    const float* x      = (const float*)d_in[0];   // (512, 512) f32
    const int*   labels = (const int*)  d_in[1];   // (512,) i32
    const float* W      = (const float*)d_in[2];   // (3, 50000, 512) f32

    float* wf   = (float*)d_out;                       // (512, 50000) f32
    float* loss = wf + (size_t)NBATCH * OUT_F;         // scalar, appended

    _Float16* xn = (_Float16*)d_ws;                                     // 512*512 f16
    float* wsL = (float*)((char*)d_ws + (size_t)NBATCH * IN_F * 2);     // 512 f32
    float* wsG = wsL + NBATCH;                                          // 9 f32

    k_normalize<<<NBATCH, 256, 0, stream>>>(x, xn);

    dim3 g2(4, (OUT_F + 127) / 128);   // N fast-moving so O-blocks share W via L2
    k_gemm_max<<<g2, 256, 0, stream>>>(xn, W, wf);

    k_rowloss<<<NBATCH, 256, 0, stream>>>(wf, labels, wsL);
    k_gram<<<9, 256, 0, stream>>>(W, wsG);
    k_final<<<1, 256, 0, stream>>>(wsL, wsG, loss);
}
